// _PointWebSAModuleBase_5523327942831
// MI455X (gfx1250) — matrix operand order repaired and checked by reading, not compiled, not run on hardware
//
#include <hip/hip_runtime.h>
#include <hip/hip_bf16.h>

#define NPTS 16384
#define MPTS 1024
#define CFEAT 64
#define SAMP 16
#define WPB 2   // waves (groups) per block in the group kernel

typedef __attribute__((ext_vector_type(16))) _Float16 v16h;
typedef __attribute__((ext_vector_type(8)))  float    v8f;

union Frag { v16h h; unsigned u[8]; };

// ---- WMMA helpers -------------------------------------------------------

__device__ __forceinline__ v8f wmma16(v16h a, v16h b, v8f c) {
  // D = A(16x32 f16) * B(32x16 f16) + C(16x16 f32)
  return __builtin_amdgcn_wmma_f32_16x16x32_f16(
      /*neg_a=*/false, a, /*neg_b=*/false, b,
      /*c_mod=*/(short)0, c, /*reuse_a=*/false, /*reuse_b=*/false);
}

// A-matrix 16-bit K packing (ISA 7.12.2): lanes0-15: v0..3 -> K 0..7, v4..7 -> K 16..23
// lanes16-31: +8
__device__ __forceinline__ int kmapA(int v, int hi) {
  return ((v < 4) ? (2 * v) : (8 + 2 * v)) + 8 * hi;
}
// B-matrix 16-bit K packing (same as A): lanes0-15: v0..3 -> K 0..7, v4..7 -> K 16..23; lanes16-31: +8
__device__ __forceinline__ int kmapB(int v, int hi) {
  return ((v < 4) ? (2 * v) : (8 + 2 * v)) + 8 * hi;
}

__device__ __forceinline__ unsigned pkf16(float a, float b) {
  _Float16 ha = (_Float16)a, hb = (_Float16)b;
  unsigned short ua = __builtin_bit_cast(unsigned short, ha);
  unsigned short ub = __builtin_bit_cast(unsigned short, hb);
  return (unsigned)ua | ((unsigned)ub << 16);
}

// ---- Kernel 1: furthest point sampling (one block per batch) ------------

__global__ void fps_kernel(const float* __restrict__ xyz, float* __restrict__ new_xyz) {
  const int b = blockIdx.x, tid = threadIdx.x;
  __shared__ float s_p[3];
  __shared__ float s_val[32];
  __shared__ int   s_idx[32];
  __shared__ int   s_last;

  float md[16];
#pragma unroll
  for (int t = 0; t < 16; ++t) md[t] = 1e10f;

  if (tid == 0) {
    new_xyz[(b * MPTS + 0) * 3 + 0] = xyz[(size_t)b * NPTS * 3 + 0];
    new_xyz[(b * MPTS + 0) * 3 + 1] = xyz[(size_t)b * NPTS * 3 + 1];
    new_xyz[(b * MPTS + 0) * 3 + 2] = xyz[(size_t)b * NPTS * 3 + 2];
  }
  int last = 0;
  for (int it = 1; it < MPTS; ++it) {
    if (tid < 3) s_p[tid] = xyz[((size_t)b * NPTS + last) * 3 + tid];
    __syncthreads();
    const float px = s_p[0], py = s_p[1], pz = s_p[2];
    float best = -1.0f; int bidx = 0;
#pragma unroll
    for (int t = 0; t < 16; ++t) {
      const int n = t * 1024 + tid;
      const float* q = &xyz[((size_t)b * NPTS + n) * 3];
      const float dx = q[0] - px, dy = q[1] - py, dz = q[2] - pz;
      const float d = dx * dx + dy * dy + dz * dz;
      md[t] = fminf(md[t], d);
      if (md[t] > best) { best = md[t]; bidx = n; }
    }
    // wave32 argmax (first-index tie-break)
#pragma unroll
    for (int off = 1; off < 32; off <<= 1) {
      const float ov = __shfl_xor(best, off, 32);
      const int   oi = __shfl_xor(bidx, off, 32);
      if (ov > best || (ov == best && oi < bidx)) { best = ov; bidx = oi; }
    }
    const int wid = tid >> 5;
    if ((tid & 31) == 0) { s_val[wid] = best; s_idx[wid] = bidx; }
    __syncthreads();
    if (tid < 32) {
      best = s_val[tid]; bidx = s_idx[tid];
#pragma unroll
      for (int off = 1; off < 32; off <<= 1) {
        const float ov = __shfl_xor(best, off, 32);
        const int   oi = __shfl_xor(bidx, off, 32);
        if (ov > best || (ov == best && oi < bidx)) { best = ov; bidx = oi; }
      }
      if (tid == 0) {
        s_last = bidx;
        new_xyz[(b * MPTS + it) * 3 + 0] = xyz[((size_t)b * NPTS + bidx) * 3 + 0];
        new_xyz[(b * MPTS + it) * 3 + 1] = xyz[((size_t)b * NPTS + bidx) * 3 + 1];
        new_xyz[(b * MPTS + it) * 3 + 2] = xyz[((size_t)b * NPTS + bidx) * 3 + 2];
      }
    }
    __syncthreads();
    last = s_last;
  }
}

// ---- Kernel 2: pack weights into WMMA B-fragment layout -----------------
// Fragments: Wa1 (32x67): 2 ntiles x 3 ktiles -> [0,6)
//            W1  (64x67): 4 x 3              -> [6,18)
//            W2  (64x64): 4 x 2              -> [18,26)
//            W3  (128x64): 8 x 2             -> [26,42)
// Each fragment: 32 lanes x 8 dwords, lane-major.

__global__ void pack_weights(const float* __restrict__ Wa1, const float* __restrict__ W1,
                             const float* __restrict__ W2, const float* __restrict__ W3,
                             unsigned* __restrict__ frags) {
  const int gid = blockIdx.x * blockDim.x + threadIdx.x;
  if (gid >= 42 * 32) return;
  const int f = gid >> 5, lane = gid & 31;
  const int n16 = lane & 15, hi = lane >> 4;
  const float* W; int K, nt, kt;
  if (f < 6)        { W = Wa1; K = 67; nt = f / 3;        kt = f % 3; }
  else if (f < 18)  { W = W1;  K = 67; nt = (f - 6) / 3;  kt = (f - 6) % 3; }
  else if (f < 26)  { W = W2;  K = 64; nt = (f - 18) / 2; kt = (f - 18) % 2; }
  else              { W = W3;  K = 64; nt = (f - 26) / 2; kt = (f - 26) % 2; }
  const int n = nt * 16 + n16;
  unsigned* out = frags + ((size_t)(f * 32 + lane)) * 8;
#pragma unroll
  for (int v = 0; v < 8; ++v) {
    const int k0 = kt * 32 + kmapB(v, hi);
    const float a = (k0 < K)     ? W[n * K + k0]     : 0.0f;
    const float c = (k0 + 1 < K) ? W[n * K + k0 + 1] : 0.0f;
    out[v] = pkf16(a, c);
  }
}

// ---- Kernel 3: brute-force kNN (one thread per query) -------------------

__global__ void knn_kernel(const float* __restrict__ xyz, const float* __restrict__ new_xyz,
                           int* __restrict__ knn_idx) {
  const int g = blockIdx.x * blockDim.x + threadIdx.x;  // 0..B*M-1
  const int b = g / MPTS;
  const float qx = new_xyz[g * 3 + 0], qy = new_xyz[g * 3 + 1], qz = new_xyz[g * 3 + 2];
  float bd[SAMP]; int bi[SAMP];
#pragma unroll
  for (int k = 0; k < SAMP; ++k) { bd[k] = 3e38f; bi[k] = 0; }
  const float* base = xyz + (size_t)b * NPTS * 3;
  for (int n = 0; n < NPTS; ++n) {
    const float dx = base[n * 3 + 0] - qx;
    const float dy = base[n * 3 + 1] - qy;
    const float dz = base[n * 3 + 2] - qz;
    const float d = dx * dx + dy * dy + dz * dz;
    if (d < bd[SAMP - 1]) {
      bd[SAMP - 1] = d; bi[SAMP - 1] = n;
#pragma unroll
      for (int k = SAMP - 1; k > 0; --k) {
        if (bd[k] < bd[k - 1]) {
          const float tv = bd[k]; bd[k] = bd[k - 1]; bd[k - 1] = tv;
          const int   ti = bi[k]; bi[k] = bi[k - 1]; bi[k - 1] = ti;
        }
      }
    }
  }
#pragma unroll
  for (int k = 0; k < SAMP; ++k) knn_idx[(size_t)g * SAMP + k] = bi[k];
}

// ---- Kernel 4: gather + AFA + MLP + maxpool (one wave per group) --------

__device__ __forceinline__ Frag load_frag(const unsigned* __restrict__ base, int fragIdx, int lane) {
  Frag f;
  const unsigned* p = base + ((size_t)(fragIdx * 32 + lane)) * 8;
#pragma unroll
  for (int v = 0; v < 8; ++v) f.u[v] = p[v];
  return f;
}

__global__ void __launch_bounds__(WPB * 32)
group_afa_mlp(const float* __restrict__ xyz, const float* __restrict__ features,
              const float* __restrict__ new_xyz, const int* __restrict__ knn_idx,
              const unsigned* __restrict__ wfrag,
              const float* __restrict__ ba1, const float* __restrict__ wa2,
              const float* __restrict__ b1, const float* __restrict__ b2,
              const float* __restrict__ b3, float* __restrict__ outF) {
  __shared__ float gld[WPB][SAMP][96];   // grouped features, zero-padded K
  __shared__ float g2 [WPB][SAMP][96];   // AFA-adjusted features
  __shared__ float hb [WPB][SAMP][64];   // MLP hidden
  __shared__ float ssc[WPB][SAMP];       // AFA scores

  const int w    = threadIdx.x >> 5;
  const int lane = threadIdx.x & 31;
  const int gidx = blockIdx.x * WPB + w;           // 0..B*M-1
  const int b = gidx / MPTS, m = gidx % MPTS;
  const int mrow = lane & 15, hi = lane >> 4;

  // ---- gather g = [xyz_rel | feat] into LDS, zero pad 67..95 ----
  const int* idxp = knn_idx + (size_t)gidx * SAMP;
  const float nx = new_xyz[gidx * 3 + 0];
  const float ny = new_xyz[gidx * 3 + 1];
  const float nz = new_xyz[gidx * 3 + 2];
  for (int t = lane; t < SAMP * 96; t += 32) {
    const int s = t / 96, c = t % 96;
    float v = 0.0f;
    const int id = idxp[s];
    if (c < 3) {
      const float sub = (c == 0) ? nx : ((c == 1) ? ny : nz);
      v = xyz[((size_t)b * NPTS + id) * 3 + c] - sub;
    } else if (c < 67) {
      v = features[((size_t)b * CFEAT + (c - 3)) * NPTS + id];
    }
    gld[w][s][c] = v;
  }
  __builtin_amdgcn_wave_barrier();

  // ---- AFA: per row i, diff(16x67) -> MLP(32) -> score -> softmax -> residual ----
  const float wa2v0 = wa2[mrow], wa2v1 = wa2[16 + mrow];
  const float bb0 = ba1[mrow],  bb1 = ba1[16 + mrow];
  for (int i = 0; i < SAMP; ++i) {
    Frag A[3];
#pragma unroll
    for (int kt = 0; kt < 3; ++kt)
#pragma unroll
      for (int v = 0; v < 8; ++v) {
        const int k0 = kt * 32 + kmapA(v, hi);
        const float a0 = gld[w][i][k0]     - gld[w][mrow][k0];
        const float a1 = gld[w][i][k0 + 1] - gld[w][mrow][k0 + 1];
        A[kt].u[v] = pkf16(a0, a1);
      }
    v8f acc0, acc1;
#pragma unroll
    for (int r = 0; r < 8; ++r) { acc0[r] = bb0; acc1[r] = bb1; }
#pragma unroll
    for (int kt = 0; kt < 3; ++kt) {
      Frag B0 = load_frag(wfrag, 0 * 3 + kt, lane);
      acc0 = wmma16(A[kt].h, B0.h, acc0);
      Frag B1 = load_frag(wfrag, 1 * 3 + kt, lane);
      acc1 = wmma16(A[kt].h, B1.h, acc1);
    }
    // relu + wa2 dot, reduce over the 16 lanes of each half (cols of the tile)
    float p[8];
#pragma unroll
    for (int r = 0; r < 8; ++r)
      p[r] = fmaxf(acc0[r], 0.0f) * wa2v0 + fmaxf(acc1[r], 0.0f) * wa2v1;
#pragma unroll
    for (int off = 1; off < 16; off <<= 1)
#pragma unroll
      for (int r = 0; r < 8; ++r) p[r] += __shfl_xor(p[r], off, 32);
    if (mrow == 0) {
#pragma unroll
      for (int r = 0; r < 8; ++r) ssc[w][hi * 8 + r] = p[r];  // score[j = 8*hi + r]
    }
    __builtin_amdgcn_wave_barrier();
    // softmax over j (redundant per lane)
    float sc[SAMP]; float mx = -3e38f;
#pragma unroll
    for (int j = 0; j < SAMP; ++j) { sc[j] = ssc[w][j]; mx = fmaxf(mx, sc[j]); }
    float sum = 0.0f;
#pragma unroll
    for (int j = 0; j < SAMP; ++j) { sc[j] = __expf(sc[j] - mx); sum += sc[j]; }
    const float inv = 1.0f / sum;
    // g2[i] = 2*g[i] - sum_j w_j * g[j]   (channels c = lane, lane+32, lane+64)
#pragma unroll
    for (int cc = 0; cc < 3; ++cc) {
      const int c = lane + cc * 32;
      float s1 = 0.0f;
#pragma unroll
      for (int j = 0; j < SAMP; ++j) s1 += sc[j] * gld[w][j][c];
      g2[w][i][c] = 2.0f * gld[w][i][c] - s1 * inv;
    }
    __builtin_amdgcn_wave_barrier();
  }

  // ---- MLP stage 1: h1 = relu(g2 @ W1^T + b1)  (16x67 -> 16x64) ----
  Frag Ag[3];
#pragma unroll
  for (int kt = 0; kt < 3; ++kt)
#pragma unroll
    for (int v = 0; v < 8; ++v) {
      const int k0 = kt * 32 + kmapA(v, hi);
      Ag[kt].u[v] = pkf16(g2[w][mrow][k0], g2[w][mrow][k0 + 1]);
    }
#pragma unroll
  for (int nt = 0; nt < 4; ++nt) {
    v8f acc; const float bv = b1[nt * 16 + mrow];
#pragma unroll
    for (int r = 0; r < 8; ++r) acc[r] = bv;
#pragma unroll
    for (int kt = 0; kt < 3; ++kt) {
      Frag Bf = load_frag(wfrag, 6 + nt * 3 + kt, lane);
      acc = wmma16(Ag[kt].h, Bf.h, acc);
    }
#pragma unroll
    for (int r = 0; r < 8; ++r) hb[w][hi * 8 + r][nt * 16 + mrow] = fmaxf(acc[r], 0.0f);
  }
  __builtin_amdgcn_wave_barrier();

  // ---- MLP stage 2: h2 = relu(h1 @ W2^T + b2)  (16x64 -> 16x64) ----
  Frag Ah[2];
#pragma unroll
  for (int kt = 0; kt < 2; ++kt)
#pragma unroll
    for (int v = 0; v < 8; ++v) {
      const int k0 = kt * 32 + kmapA(v, hi);
      Ah[kt].u[v] = pkf16(hb[w][mrow][k0], hb[w][mrow][k0 + 1]);
    }
  v8f acc2[4];
#pragma unroll
  for (int nt = 0; nt < 4; ++nt) {
    const float bv = b2[nt * 16 + mrow];
#pragma unroll
    for (int r = 0; r < 8; ++r) acc2[nt][r] = bv;
#pragma unroll
    for (int kt = 0; kt < 2; ++kt) {
      Frag Bf = load_frag(wfrag, 18 + nt * 2 + kt, lane);
      acc2[nt] = wmma16(Ah[kt].h, Bf.h, acc2[nt]);
    }
  }
  __builtin_amdgcn_wave_barrier();
#pragma unroll
  for (int nt = 0; nt < 4; ++nt)
#pragma unroll
    for (int r = 0; r < 8; ++r) hb[w][hi * 8 + r][nt * 16 + mrow] = fmaxf(acc2[nt][r], 0.0f);
  __builtin_amdgcn_wave_barrier();

  // ---- MLP stage 3: h3 = relu(h2 @ W3^T + b3) (16x64 -> 16x128), max over samples ----
  Frag A3[2];
#pragma unroll
  for (int kt = 0; kt < 2; ++kt)
#pragma unroll
    for (int v = 0; v < 8; ++v) {
      const int k0 = kt * 32 + kmapA(v, hi);
      A3[kt].u[v] = pkf16(hb[w][mrow][k0], hb[w][mrow][k0 + 1]);
    }
#pragma unroll
  for (int nt = 0; nt < 8; ++nt) {
    v8f acc; const float bv = b3[nt * 16 + mrow];
#pragma unroll
    for (int r = 0; r < 8; ++r) acc[r] = bv;
#pragma unroll
    for (int kt = 0; kt < 2; ++kt) {
      Frag Bf = load_frag(wfrag, 26 + nt * 2 + kt, lane);
      acc = wmma16(A3[kt].h, Bf.h, acc);
    }
    float mxv = 0.0f;  // relu(x) >= 0, so max starts at 0
#pragma unroll
    for (int r = 0; r < 8; ++r) mxv = fmaxf(mxv, acc[r]);
    mxv = fmaxf(mxv, __shfl_xor(mxv, 16, 32));   // combine rows 0-7 with 8-15
    if (hi == 0) {
      const int ch = nt * 16 + mrow;
      outF[((size_t)b * 128 + ch) * MPTS + m] = mxv;
    }
  }
}

// ---- launch -------------------------------------------------------------

extern "C" void kernel_launch(void* const* d_in, const int* in_sizes, int n_in,
                              void* d_out, int out_size, void* d_ws, size_t ws_size,
                              hipStream_t stream) {
  (void)n_in; (void)out_size; (void)ws_size;
  const float* xyz      = (const float*)d_in[0];
  const float* features = (const float*)d_in[1];
  const float* Wa1      = (const float*)d_in[2];
  const float* ba1      = (const float*)d_in[3];
  const float* wa2      = (const float*)d_in[4];
  const float* W1       = (const float*)d_in[5];
  const float* b1       = (const float*)d_in[6];
  const float* W2       = (const float*)d_in[7];
  const float* b2       = (const float*)d_in[8];
  const float* W3       = (const float*)d_in[9];
  const float* b3       = (const float*)d_in[10];
  const int B = in_sizes[0] / (NPTS * 3);

  float* new_xyz = (float*)d_out;                       // (B, M, 3)
  float* outF    = (float*)d_out + (size_t)B * MPTS * 3; // (B, 128, M)

  int* knn_idx   = (int*)d_ws;                                   // B*M*16 ints
  unsigned* wfrag = (unsigned*)((char*)d_ws + (size_t)B * MPTS * SAMP * sizeof(int));

  fps_kernel<<<B, 1024, 0, stream>>>(xyz, new_xyz);
  pack_weights<<<6, 256, 0, stream>>>(Wa1, W1, W2, W3, wfrag);
  knn_kernel<<<(B * MPTS) / 256, 256, 0, stream>>>(xyz, new_xyz, knn_idx);
  group_afa_mlp<<<(B * MPTS) / WPB, WPB * 32, 0, stream>>>(
      xyz, features, new_xyz, knn_idx, wfrag, ba1, wa2, b1, b2, b3, outF);
}
